// EnhancedSubtractionUnit_45835890983317
// MI455X (gfx1250) — matrix operand order repaired and checked by reading, not compiled, not run on hardware
//
#include <hip/hip_runtime.h>
#include <cstdint>

typedef __attribute__((ext_vector_type(16))) _Float16 v16h;
typedef __attribute__((ext_vector_type(8)))  _Float16 v8h;
typedef __attribute__((ext_vector_type(8)))  float    v8f;

// ---------------------------------------------------------------------------
// Shapes: B=8, C=256, H=W=64, r=64.  spatial = 4096 = 1<<12.
// Activations feeding WMMA convs are NHWC f16; everything else NCHW f32.
// GEMM K ordering: k = r*CIN + ci   (3x3 tap outer, channel inner)
// ---------------------------------------------------------------------------

// ---- pack x_low||x_high into NHWC f16 [8,64,64,512] ----
__global__ __launch_bounds__(256)
void k_pack_cat(const float* __restrict__ xl, const float* __restrict__ xh,
                _Float16* __restrict__ cat) {
  const int i = blockIdx.x * 256 + threadIdx.x;   // < 8*4096*512 = 2^24
  const int c = i & 511;
  const int pix = i >> 9;
  const int b = pix >> 12;
  const int s = pix & 4095;
  const float v = (c < 256) ? xl[(((b << 8) + c) << 12) + s]
                            : xh[(((b << 8) + (c - 256)) << 12) + s];
  cat[i] = (_Float16)v;
}

// ---- repack conv weights [co][ci][3][3] f32 -> [co][r*CIN+ci] f16 ----
__global__ __launch_bounds__(256)
void k_repack_w(const float* __restrict__ src, _Float16* __restrict__ dst,
                int CIN, int total) {
  const int i = blockIdx.x * 256 + threadIdx.x;
  if (i >= total) return;
  const int K = CIN * 9;
  const int co = i / K;
  const int rem = i - co * K;
  const int r = rem / CIN;
  const int ci = rem - r * CIN;
  dst[i] = (_Float16)src[(size_t)(co * CIN + ci) * 9 + r];
}

// ---------------------------------------------------------------------------
// Implicit-GEMM 3x3 SAME conv via WMMA.  M=256 outch, N=32768 pixels, K=CIN*9.
// Block = 256 threads = 8 waves laid out 4(M) x 2(N); per-wave 32x64 output
// (2x4 subtiles of 16x16).  LDS: A[m][k], B[n][k] (both k-contiguous).
// NHWC input => B tile staging is ONE conditional 32B load per thread.
// Register double-buffering hides global latency behind the 8 WMMAs.
// ---------------------------------------------------------------------------
template <int CIN>
__global__ __launch_bounds__(256)
void k_conv3x3_wmma(const _Float16* __restrict__ xin,   // NHWC [8,64,64,CIN] f16
                    const _Float16* __restrict__ wgt,   // [256][CIN*9] f16, k-reordered
                    const float* __restrict__ bias,     // [256]
                    float* __restrict__ out_f32,        // optional
                    _Float16* __restrict__ out_f16,     // optional, NCHW
                    const float* __restrict__ bn_g, const float* __restrict__ bn_b,
                    const float* __restrict__ bn_m, const float* __restrict__ bn_v,
                    int do_bnrelu) {
  constexpr int K  = CIN * 9;
  constexpr int L2 = (CIN == 512) ? 9 : 8;     // log2(CIN)
  __shared__ alignas(32) _Float16 As[128 * 32];
  __shared__ alignas(32) _Float16 Bs[128 * 32];

  const int t    = threadIdx.x;
  const int lane = t & 31;
  const int wave = t >> 5;
  const int wm   = wave >> 1;       // 0..3  (M sub-block of 32)
  const int wn   = wave & 1;        // 0..1  (N sub-block of 64)
  const int lr   = lane & 15;
  const int hi   = lane >> 4;       // 0 / 1

  const int nbase = blockIdx.x * 128;
  const int mbase = blockIdx.y * 128;

  // loader mapping: 2 threads per row, 16 halves each
  const int lrow = t >> 1;          // 0..127
  const int kseg = (t & 1) << 4;    // 0 or 16
  const int n_g  = nbase + lrow;
  const int bb   = n_g >> 12;
  const int yy0  = (n_g >> 6) & 63;
  const int xx0  = n_g & 63;

  const _Float16* wrow = wgt + (size_t)(mbase + lrow) * K + kseg;
  const v8h zero8 = {};

  v8h ra0, ra1, rb0, rb1;
  auto load_tile = [&](int kb) {
    // A: contiguous weight row
    const v8h* sa = reinterpret_cast<const v8h*>(wrow + kb);
    ra0 = sa[0];
    ra1 = sa[1];
    __builtin_prefetch(wrow + kb + 64, 0, 3);
    // B: 16 consecutive channels at one shifted pixel (single bounds check)
    const int kg  = kb + kseg;
    const int r   = kg >> L2;              // 0..8
    const int ci0 = kg & (CIN - 1);
    const int dy  = r / 3;
    const int dx  = r - dy * 3;
    const int yy  = yy0 + dy - 1;
    const int xx  = xx0 + dx - 1;
    rb0 = zero8;
    rb1 = zero8;
    if ((unsigned)yy < 64u && (unsigned)xx < 64u) {
      const v8h* sb = reinterpret_cast<const v8h*>(
          xin + (((size_t)((bb << 12) + (yy << 6) + xx)) << L2) + ci0);
      rb0 = sb[0];
      rb1 = sb[1];
    }
  };

  v8f acc[2][4] = {};
  load_tile(0);

  for (int kb = 0; kb < K; kb += 32) {
    // ---- commit staged registers to LDS ----
    {
      v8h* da = reinterpret_cast<v8h*>(&As[lrow * 32 + kseg]);
      da[0] = ra0;
      da[1] = ra1;
      v8h* db2 = reinterpret_cast<v8h*>(&Bs[lrow * 32 + kseg]);
      db2[0] = rb0;
      db2[1] = rb1;
    }
    __syncthreads();

    // ---- issue next tile's global loads (latency hidden by WMMAs) ----
    if (kb + 32 < K) load_tile(kb + 32);

    // ---- fragments (A and B both hold k = 16*hi + s, contiguous order) ----
    v16h afrag[2];
    #pragma unroll
    for (int mi = 0; mi < 2; ++mi) {
      const _Float16* base = &As[(wm * 32 + mi * 16 + lr) * 32];
      const int k0 = hi ? 16 : 0;          // lanes>=16 hold K 16..31
      v8h lo = *reinterpret_cast<const v8h*>(base + k0);
      v8h h8 = *reinterpret_cast<const v8h*>(base + k0 + 8);
      v16h a;
      #pragma unroll
      for (int j = 0; j < 8; ++j) { a[j] = lo[j]; a[j + 8] = h8[j]; }
      afrag[mi] = a;
    }
    v16h bfrag[4];
    #pragma unroll
    for (int ni = 0; ni < 4; ++ni) {
      const _Float16* base = &Bs[(wn * 64 + ni * 16 + lr) * 32 + (hi ? 16 : 0)];
      bfrag[ni] = *reinterpret_cast<const v16h*>(base);  // sequential K 0..15 / 16..31
    }

    #pragma unroll
    for (int mi = 0; mi < 2; ++mi)
      #pragma unroll
      for (int ni = 0; ni < 4; ++ni)
        acc[mi][ni] = __builtin_amdgcn_wmma_f32_16x16x32_f16(
            false, afrag[mi], false, bfrag[ni], (short)0, acc[mi][ni], false, false);
    __syncthreads();
  }

  // ---- epilogue: bias (+ optional BN+ReLU), write f32 / f16 NCHW ----
  #pragma unroll
  for (int mi = 0; mi < 2; ++mi) {
    #pragma unroll
    for (int ni = 0; ni < 4; ++ni) {
      const int n  = nbase + wn * 64 + ni * 16 + lr;
      const int b  = n >> 12;
      const int sp = n & 4095;
      #pragma unroll
      for (int j = 0; j < 8; ++j) {
        const int co = mbase + wm * 32 + mi * 16 + (hi ? 8 : 0) + j;
        float y = acc[mi][ni][j] + bias[co];
        if (do_bnrelu) {
          y = bn_g[co] * (y - bn_m[co]) * rsqrtf(bn_v[co] + 1e-5f) + bn_b[co];
          y = fmaxf(y, 0.f);
        }
        const size_t addr = (((size_t)b * 256 + co) << 12) + sp;
        if (out_f32) out_f32[addr] = y;
        if (out_f16) out_f16[addr] = (_Float16)y;
      }
    }
  }
}

// ---- tiny 3x3 conv 256->2 (offset prediction), reads NCHW h16 ----
__global__ __launch_bounds__(256)
void k_offset_conv(const _Float16* __restrict__ h, const float* __restrict__ a2w,
                   const float* __restrict__ a2b, float* __restrict__ offs) {
  __shared__ float w[2 * 2304];
  for (int i = threadIdx.x; i < 2 * 2304; i += 256) w[i] = a2w[i];
  __syncthreads();
  const int n = blockIdx.x * 256 + threadIdx.x;  // 0..32767
  const int b = n >> 12, s = n & 4095, y = s >> 6, x = s & 63;
  float a0 = a2b[0], a1 = a2b[1];
  for (int ci = 0; ci < 256; ++ci) {
    const _Float16* base = h + (((size_t)b * 256 + ci) << 12);
    #pragma unroll
    for (int r = 0; r < 9; ++r) {
      const int yy = y + r / 3 - 1, xx = x + r % 3 - 1;
      if ((unsigned)yy < 64u && (unsigned)xx < 64u) {
        const float v = (float)base[(yy << 6) + xx];
        a0 += v * w[ci * 9 + r];
        a1 += v * w[2304 + ci * 9 + r];
      }
    }
  }
  offs[((b * 2 + 0) << 12) + s] = a0;
  offs[((b * 2 + 1) << 12) + s] = a1;
}

// ---- grid build + bilinear warp + diff (f32 NCHW and f16 NHWC copies) ----
__global__ __launch_bounds__(256)
void k_warp_diff(const float* __restrict__ xl, const float* __restrict__ xh,
                 const float* __restrict__ offs,
                 float* __restrict__ diff, _Float16* __restrict__ diff16) {
  const int i = blockIdx.x * 256 + threadIdx.x;   // < 8*256*4096
  const int s = i & 4095, c = (i >> 12) & 255, b = i >> 20;
  const int y = s >> 6, x = s & 63;
  // reference: y-linspace sits in grid ch0 (drives ix), x-linspace in ch1 (drives iy)
  const float g0 = -1.f + 2.f * (float)y * (1.f / 63.f) + offs[((b * 2 + 0) << 12) + s];
  const float g1 = -1.f + 2.f * (float)x * (1.f / 63.f) + offs[((b * 2 + 1) << 12) + s];
  const float ix = (g0 + 1.f) * 32.f - 0.5f;
  const float iy = (g1 + 1.f) * 32.f - 0.5f;
  const float x0f = floorf(ix), y0f = floorf(iy);
  const float wx = ix - x0f, wy = iy - y0f;
  const int x0 = (int)x0f, y0 = (int)y0f;
  const float* img = xh + (((size_t)b * 256 + c) << 12);
  auto gat = [&](int yy, int xx) -> float {
    return ((unsigned)yy < 64u && (unsigned)xx < 64u) ? img[(yy << 6) + xx] : 0.f;
  };
  const float v00 = gat(y0, x0),     v01 = gat(y0, x0 + 1);
  const float v10 = gat(y0 + 1, x0), v11 = gat(y0 + 1, x0 + 1);
  const float samp = v00 * (1.f - wx) * (1.f - wy) + v01 * wx * (1.f - wy)
                   + v10 * (1.f - wx) * wy + v11 * wx * wy;
  const float d = xl[i] - samp;
  diff[i] = d;                                   // NCHW f32
  diff16[(((size_t)b << 12) + s) * 256 + c] = (_Float16)d;  // NHWC f16
}

// ---- global average pool per (b, co) over NCHW xd ----
__global__ __launch_bounds__(256)
void k_pool(const float* __restrict__ xd, float* __restrict__ pooled) {
  __shared__ float red[256];
  const float* src = xd + ((size_t)blockIdx.x << 12);
  float sum = 0.f;
  for (int i = threadIdx.x; i < 4096; i += 256) sum += src[i];
  red[threadIdx.x] = sum;
  __syncthreads();
  for (int st = 128; st > 0; st >>= 1) {
    if (threadIdx.x < st) red[threadIdx.x] += red[threadIdx.x + st];
    __syncthreads();
  }
  if (threadIdx.x == 0) pooled[blockIdx.x] = red[0] * (1.f / 4096.f);
}

// ---- SE MLP: 256 -> relu(64) -> sigmoid(256), one block per batch ----
__global__ __launch_bounds__(256)
void k_se(const float* __restrict__ pooled,
          const float* __restrict__ w1, const float* __restrict__ b1,
          const float* __restrict__ w2, const float* __restrict__ b2,
          float* __restrict__ s) {
  __shared__ float p[256];
  __shared__ float hbuf[64];
  const int b = blockIdx.x, t = threadIdx.x;
  p[t] = pooled[b * 256 + t];
  __syncthreads();
  if (t < 64) {
    float a = b1[t];
    for (int c = 0; c < 256; ++c) a += w1[t * 256 + c] * p[c];
    hbuf[t] = fmaxf(a, 0.f);
  }
  __syncthreads();
  float a = b2[t];
  for (int j = 0; j < 64; ++j) a += w2[t * 64 + j] * hbuf[j];
  s[b * 256 + t] = 1.f / (1.f + expf(-a));
}

__global__ __launch_bounds__(256)
void k_fuse(const float* __restrict__ xd, const float* __restrict__ s,
            float* __restrict__ fused, int first) {
  const int i = blockIdx.x * 256 + threadIdx.x;
  const float v = xd[i] * s[i >> 12];
  fused[i] = first ? v : (fused[i] + v);
}

// ---- spatial attention conv 256->1 + sigmoid (NCHW fused) ----
__global__ __launch_bounds__(256)
void k_attn(const float* __restrict__ fused, const float* __restrict__ saw,
            const float* __restrict__ sab, float* __restrict__ attn) {
  __shared__ float w[2304];
  for (int i = threadIdx.x; i < 2304; i += 256) w[i] = saw[i];
  __syncthreads();
  const int n = blockIdx.x * 256 + threadIdx.x;
  const int b = n >> 12, s = n & 4095, y = s >> 6, x = s & 63;
  float a = sab[0];
  for (int ci = 0; ci < 256; ++ci) {
    const float* base = fused + (((size_t)b * 256 + ci) << 12);
    #pragma unroll
    for (int r = 0; r < 9; ++r) {
      const int yy = y + r / 3 - 1, xx = x + r % 3 - 1;
      if ((unsigned)yy < 64u && (unsigned)xx < 64u)
        a += base[(yy << 6) + xx] * w[ci * 9 + r];
    }
  }
  attn[n] = 1.f / (1.f + expf(-a));
}

__global__ __launch_bounds__(256)
void k_final(const float* __restrict__ attn, const float* __restrict__ diff,
             const float* __restrict__ xl, float* __restrict__ out) {
  const int i = blockIdx.x * 256 + threadIdx.x;
  const int b = i >> 20, s = i & 4095;
  out[i] = attn[(b << 12) + s] * diff[i] + xl[i];
}

// ---------------------------------------------------------------------------
extern "C" void kernel_launch(void* const* d_in, const int* in_sizes, int n_in,
                              void* d_out, int out_size, void* d_ws, size_t ws_size,
                              hipStream_t stream) {
  (void)in_sizes; (void)n_in; (void)out_size; (void)ws_size;
  const float* x_low = (const float*)d_in[0];
  const float* x_high = (const float*)d_in[1];
  const float* a1w = (const float*)d_in[2];
  const float* a1b = (const float*)d_in[3];
  const float* bn_g = (const float*)d_in[4];
  const float* bn_b = (const float*)d_in[5];
  const float* bn_m = (const float*)d_in[6];
  const float* bn_v = (const float*)d_in[7];
  const float* a2w = (const float*)d_in[8];
  const float* a2b = (const float*)d_in[9];
  const float* dw  = (const float*)d_in[10];
  const float* db  = (const float*)d_in[11];
  const float* se1w = (const float*)d_in[12];
  const float* se1b = (const float*)d_in[13];
  const float* se2w = (const float*)d_in[14];
  const float* se2b = (const float*)d_in[15];
  const float* saw = (const float*)d_in[16];
  const float* sab = (const float*)d_in[17];
  float* out = (float*)d_out;

  char* ws = (char*)d_ws;
  size_t off = 0;
  auto alloc = [&](size_t bytes) {
    char* p = ws + off;
    off = (off + bytes + 255) & ~(size_t)255;
    return p;
  };
  const size_t NPIX = (size_t)8 * 4096;     // 32768
  const size_t NELT = NPIX * 256;           // 8388608

  _Float16* cat16  = (_Float16*)alloc(NPIX * 512 * sizeof(_Float16)); // NHWC
  _Float16* a1w16  = (_Float16*)alloc((size_t)256 * 4608 * sizeof(_Float16));
  _Float16* dw16   = (_Float16*)alloc((size_t)3 * 256 * 2304 * sizeof(_Float16));
  _Float16* h16    = (_Float16*)alloc(NELT * sizeof(_Float16));       // NCHW
  float*    offs   = (float*)alloc(NPIX * 2 * sizeof(float));
  float*    diff   = (float*)alloc(NELT * sizeof(float));             // NCHW
  _Float16* diff16 = (_Float16*)alloc(NELT * sizeof(_Float16));       // NHWC
  float*    xd     = (float*)alloc(NELT * sizeof(float));             // NCHW
  float*    fused  = (float*)alloc(NELT * sizeof(float));             // NCHW
  float*    pooled = (float*)alloc(2048 * sizeof(float));
  float*    svec   = (float*)alloc(2048 * sizeof(float));
  float*    attn   = (float*)alloc(NPIX * sizeof(float));

  // precision packing / weight K-reordering
  k_pack_cat<<<(8u * 512 * 4096) / 256, 256, 0, stream>>>(x_low, x_high, cat16);
  k_repack_w<<<(256 * 4608 + 255) / 256, 256, 0, stream>>>(a1w, a1w16, 512, 256 * 4608);
  k_repack_w<<<(768 * 2304 + 255) / 256, 256, 0, stream>>>(dw, dw16, 256, 768 * 2304);

  const dim3 gconv(32768 / 128, 256 / 128);   // N tiles x M tiles

  // conv1 (512->256) + BN + ReLU -> h16 (NCHW f16)
  k_conv3x3_wmma<512><<<gconv, 256, 0, stream>>>(cat16, a1w16, a1b, nullptr, h16,
                                                 bn_g, bn_b, bn_m, bn_v, 1);
  // offset prediction + warp + diff
  k_offset_conv<<<32768 / 256, 256, 0, stream>>>(h16, a2w, a2b, offs);
  k_warp_diff<<<(unsigned)(NELT / 256), 256, 0, stream>>>(x_low, x_high, offs, diff, diff16);

  // three dynamic-scale branches
  for (int k = 0; k < 3; ++k) {
    k_conv3x3_wmma<256><<<gconv, 256, 0, stream>>>(
        diff16, dw16 + (size_t)k * 256 * 2304, db + k * 256, xd, nullptr,
        nullptr, nullptr, nullptr, nullptr, 0);
    k_pool<<<2048, 256, 0, stream>>>(xd, pooled);
    k_se<<<8, 256, 0, stream>>>(pooled, se1w + k * 64 * 256, se1b + k * 64,
                                se2w + k * 256 * 64, se2b + k * 256, svec);
    k_fuse<<<(unsigned)(NELT / 256), 256, 0, stream>>>(xd, svec, fused, (k == 0) ? 1 : 0);
  }

  // spatial attention + final blend
  k_attn<<<32768 / 256, 256, 0, stream>>>(fused, saw, sab, attn);
  k_final<<<(unsigned)(NELT / 256), 256, 0, stream>>>(attn, diff, x_low, out);
}
